// TransformerBlock_30666066493654
// MI455X (gfx1250) — matrix operand order repaired and checked by reading, not compiled, not run on hardware
//
#include <hip/hip_runtime.h>
#include <cstdint>

typedef __bf16 bf16_t;
typedef __attribute__((ext_vector_type(16))) __bf16 v16bf;
typedef __attribute__((ext_vector_type(8)))  float  v8f;

#define B_    4
#define NSEQ  2048
#define CDIM  384
#define HEADS 6
#define DHEAD 64
#define HIDD  1536
#define MROWS (B_ * NSEQ)
#define QSZ   ((size_t)B_ * HEADS * NSEQ * DHEAD)

static __device__ inline v8f vzero8() {
    v8f z = {0.f, 0.f, 0.f, 0.f, 0.f, 0.f, 0.f, 0.f};
    return z;
}

// low 32 bits of a generic pointer to LDS == LDS byte offset
static __device__ inline unsigned lds_off(const void* p) {
    return (unsigned)(uintptr_t)p;
}

// async global -> LDS copy of 16 bytes, tracked by ASYNCcnt (in-order)
static __device__ inline void async_copy_b128(unsigned lds_byte_off, const void* gptr) {
    asm volatile("global_load_async_to_lds_b128 %0, %1, off"
                 :: "v"(lds_byte_off), "v"((unsigned long long)(uintptr_t)gptr)
                 : "memory");
}

// counted wait: block until at most N async ops remain outstanding
template <int N>
static __device__ inline void wait_async_le() {
    asm volatile("s_wait_asynccnt %0" :: "i"(N) : "memory");
}

// 16-lane butterfly via DPP16 (stays within each wave32 half-row)
template <int CTRL>
static __device__ inline float dpp_mov(float x) {
    return __int_as_float(
        __builtin_amdgcn_update_dpp(0, __float_as_int(x), CTRL, 0xf, 0xf, true));
}
static __device__ inline float row16_max(float x) {
    x = fmaxf(x, dpp_mov<0xB1>(x));   // quad_perm(1,0,3,2)  xor1
    x = fmaxf(x, dpp_mov<0x4E>(x));   // quad_perm(2,3,0,1)  xor2
    x = fmaxf(x, dpp_mov<0x141>(x));  // row_half_mirror     cross-quad
    x = fmaxf(x, dpp_mov<0x140>(x));  // row_mirror          cross-half
    return x;
}
static __device__ inline float row16_sum(float x) {
    x += dpp_mov<0xB1>(x);
    x += dpp_mov<0x4E>(x);
    x += dpp_mov<0x141>(x);
    x += dpp_mov<0x140>(x);
    return x;
}

union FragB16 {
    v16bf    v;
    unsigned u[8];
    bf16_t   e[16];
};

// ---------------------------------------------------------------------------
// fp32 -> bf16 weight conversion
// ---------------------------------------------------------------------------
__global__ void cvt_kernel(const float* __restrict__ in, bf16_t* __restrict__ out, int n) {
    int i = blockIdx.x * blockDim.x + threadIdx.x;
    if (i < n) out[i] = (bf16_t)in[i];
}

// ---------------------------------------------------------------------------
// LayerNorm over C=384, one row per 128-thread block, bf16 output
// ---------------------------------------------------------------------------
__global__ __launch_bounds__(128) void ln_kernel(const float* __restrict__ x,
                                                 const float* __restrict__ g,
                                                 const float* __restrict__ bb,
                                                 bf16_t* __restrict__ out) {
    const int row  = blockIdx.x;
    const int tid  = threadIdx.x;
    const int lane = tid & 31;
    const int wid  = tid >> 5;
    const float* xr = x + (size_t)row * CDIM;
    float v0 = xr[tid], v1 = xr[tid + 128], v2 = xr[tid + 256];
    float s  = v0 + v1 + v2;
    float s2 = v0 * v0 + v1 * v1 + v2 * v2;
#pragma unroll
    for (int off = 16; off > 0; off >>= 1) {
        s  += __shfl_xor(s, off, 32);
        s2 += __shfl_xor(s2, off, 32);
    }
    __shared__ float red[10];
    if (lane == 0) { red[wid] = s; red[4 + wid] = s2; }
    __syncthreads();
    if (tid == 0) {
        float ts = red[0] + red[1] + red[2] + red[3];
        float t2 = red[4] + red[5] + red[6] + red[7];
        float mu  = ts / (float)CDIM;
        float var = t2 / (float)CDIM - mu * mu;
        red[8] = mu;
        red[9] = rsqrtf(var + 1e-5f);
    }
    __syncthreads();
    float mu = red[8], rs = red[9];
    bf16_t* o = out + (size_t)row * CDIM;
    o[tid]       = (bf16_t)((v0 - mu) * rs * g[tid]       + bb[tid]);
    o[tid + 128] = (bf16_t)((v1 - mu) * rs * g[tid + 128] + bb[tid + 128]);
    o[tid + 256] = (bf16_t)((v2 - mu) * rs * g[tid + 256] + bb[tid + 256]);
}

// ---------------------------------------------------------------------------
// bf16 WMMA GEMM, 128x128 block tile, K-step 32, double-buffered async-LDS
// staging with counted ASYNCcnt waits, fused epilogues.
// MODE 0: QKV scatter to [B,H,N,D] (q scaled by 1/8)
// MODE 1: f32 out = residual + acc + bias       (proj)
// MODE 2: bf16 out = gelu(acc + bias)           (fc1)
// MODE 3: f32 out = residual + acc + bias       (fc2)
// ---------------------------------------------------------------------------
template <int MODE>
__global__ __launch_bounds__(256) void gemm_bf16_kernel(
    const bf16_t* __restrict__ A, const bf16_t* __restrict__ Bm,
    int Ndim, int Kdim,
    const float* __restrict__ bias, const float* __restrict__ resid,
    float* __restrict__ outF, bf16_t* __restrict__ outH) {
    __shared__ __align__(16) bf16_t sA[2][128 * 32];
    __shared__ __align__(16) bf16_t sB[2][32 * 128];

    const int tid   = threadIdx.x;
    const int lane  = tid & 31;
    const int wave  = tid >> 5;
    const int half  = lane >> 4;
    const int m16   = lane & 15;
    const int waveM = wave & 3;
    const int waveN = wave >> 2;
    const int m0    = blockIdx.y * 128;
    const int n0    = blockIdx.x * 128;

    const unsigned ldsA = lds_off(&sA[0][0]);
    const unsigned ldsB = lds_off(&sB[0][0]);

    // per-thread staging chunk geometry (512 x 16B chunks per tile, 2/thread)
    const int cA0 = tid, cA1 = tid + 256;
    const int rA0 = cA0 >> 2, oA0 = (cA0 & 3) * 8;
    const int rA1 = cA1 >> 2, oA1 = (cA1 & 3) * 8;
    const int rB0 = cA0 >> 4, oB0 = (cA0 & 15) * 8;
    const int rB1 = cA1 >> 4, oB1 = (cA1 & 15) * 8;

    auto issue_tile = [&](int k0, int buf) {
        const unsigned bo = (unsigned)buf * (128 * 32 * 2);
        async_copy_b128(ldsA + bo + cA0 * 16, A + (size_t)(m0 + rA0) * Kdim + k0 + oA0);
        async_copy_b128(ldsA + bo + cA1 * 16, A + (size_t)(m0 + rA1) * Kdim + k0 + oA1);
        async_copy_b128(ldsB + bo + cA0 * 16, Bm + (size_t)(k0 + rB0) * Ndim + n0 + oB0);
        async_copy_b128(ldsB + bo + cA1 * 16, Bm + (size_t)(k0 + rB1) * Ndim + n0 + oB1);
    };

    v8f acc[2][4];
#pragma unroll
    for (int i = 0; i < 2; i++)
#pragma unroll
        for (int j = 0; j < 4; j++) acc[i][j] = vzero8();

    const int TK = Kdim >> 5;
    issue_tile(0, 0);
    for (int t = 0; t < TK; ++t) {
        const int cur = t & 1;
        if (t + 1 < TK) {
            issue_tile((t + 1) * 32, cur ^ 1);   // overlap next tile behind compute
            if (t + 2 < TK)                       // L2 warm-up two tiles ahead
                __builtin_prefetch(A + (size_t)(m0 + rA0) * Kdim + (t + 2) * 32 + oA0, 0, 1);
            wait_async_le<4>();                   // only tile t's 4 ops must be done
        } else {
            wait_async_le<0>();
        }
        __syncthreads();

        FragB16 af[2];
#pragma unroll
        for (int mi = 0; mi < 2; mi++) {
            int r = waveM * 32 + mi * 16 + m16;
#pragma unroll
            for (int j = 0; j < 8; j++) {
                int kk = (j & 3) * 2 + (j >> 2) * 16 + half * 8;
                af[mi].u[j] = *reinterpret_cast<const unsigned*>(&sA[cur][r * 32 + kk]);
            }
        }
        FragB16 bfr[4];
#pragma unroll
        for (int ni = 0; ni < 4; ni++) {
            int ncol = waveN * 64 + ni * 16 + m16;
#pragma unroll
            for (int j = 0; j < 8; j++) {
                int lk = (j & 3) * 2 + (j >> 2) * 16 + half * 8;
                bfr[ni].e[2 * j]     = sB[cur][lk * 128 + ncol];
                bfr[ni].e[2 * j + 1] = sB[cur][(lk + 1) * 128 + ncol];
            }
        }
#pragma unroll
        for (int mi = 0; mi < 2; mi++)
#pragma unroll
            for (int ni = 0; ni < 4; ni++)
                acc[mi][ni] = __builtin_amdgcn_wmma_f32_16x16x32_bf16(
                    false, af[mi].v, false, bfr[ni].v, (short)0, acc[mi][ni], false, false);
        __syncthreads();  // buffer `cur` free for reuse at t+2
    }

    // epilogue
#pragma unroll
    for (int mi = 0; mi < 2; mi++)
#pragma unroll
        for (int ni = 0; ni < 4; ni++)
#pragma unroll
            for (int j = 0; j < 8; j++) {
                int row = m0 + waveM * 32 + mi * 16 + j + half * 8;
                int col = n0 + waveN * 64 + ni * 16 + m16;
                float v = acc[mi][ni][j];
                if constexpr (MODE == 0) {
                    int sec = col / CDIM;
                    int cc  = col % CDIM;
                    int hh  = cc / DHEAD, dd = cc % DHEAD;
                    int b   = row / NSEQ, n = row % NSEQ;
                    if (sec == 0) v *= 0.125f;  // q * D^-1/2
                    outH[(size_t)sec * QSZ + ((size_t)(b * HEADS + hh) * NSEQ + n) * DHEAD + dd] =
                        (bf16_t)v;
                } else if constexpr (MODE == 1 || MODE == 3) {
                    size_t idx = (size_t)row * Ndim + col;
                    outF[idx]  = resid[idx] + v + bias[col];
                } else {
                    float t = v + bias[col];
                    float ge = 0.5f * t * (1.0f + erff(t * 0.70710678118f));
                    outH[(size_t)row * Ndim + col] = (bf16_t)ge;
                }
            }
}

// ---------------------------------------------------------------------------
// Flash attention: grid (B*H, N/64), 128 threads (4 waves x 16 Q rows),
// double-buffered async K/V staging, DPP softmax reductions.
// ---------------------------------------------------------------------------
__global__ __launch_bounds__(128) void attn_kernel(const bf16_t* __restrict__ Q,
                                                   const bf16_t* __restrict__ K,
                                                   const bf16_t* __restrict__ V,
                                                   bf16_t* __restrict__ O) {
    __shared__ __align__(16) bf16_t sK[2][64 * DHEAD];
    __shared__ __align__(16) bf16_t sV[2][64 * DHEAD];
    __shared__ __align__(16) bf16_t sP[4][16 * 64];

    const int tid  = threadIdx.x;
    const int lane = tid & 31;
    const int wave = tid >> 5;
    const int half = lane >> 4;
    const int m16  = lane & 15;

    const int bh = blockIdx.x;
    const int b  = bh / HEADS;
    const int h  = bh % HEADS;
    const int q0 = blockIdx.y * 64 + wave * 16;
    const size_t base = (size_t)bh * NSEQ * DHEAD;

    const unsigned ldsK0 = lds_off(&sK[0][0]);
    const unsigned ldsV0 = lds_off(&sV[0][0]);

    auto issue_kv = [&](int kb, int buf) {
        const bf16_t* gk = K + base + (size_t)kb * 64 * DHEAD;
        const bf16_t* gv = V + base + (size_t)kb * 64 * DHEAD;
        const unsigned bo = (unsigned)buf * (64 * DHEAD * 2);
#pragma unroll
        for (int i = 0; i < 4; i++) {
            int c = tid + i * 128;
            async_copy_b128(ldsK0 + bo + c * 16, gk + c * 8);
            async_copy_b128(ldsV0 + bo + c * 16, gv + c * 8);
        }
    };

    // Q fragments (A-layout, 16 rows x 64 d, pre-scaled in QKV epilogue)
    FragB16 qf[2];
#pragma unroll
    for (int c = 0; c < 2; c++)
#pragma unroll
        for (int j = 0; j < 8; j++) {
            int kk = c * 32 + (j & 3) * 2 + (j >> 2) * 16 + half * 8;
            qf[c].u[j] =
                *reinterpret_cast<const unsigned*>(Q + base + (size_t)(q0 + m16) * DHEAD + kk);
        }

    float mrow[8], lrow[8];
    v8f acco[4];
#pragma unroll
    for (int j = 0; j < 8; j++) { mrow[j] = -3.0e38f; lrow[j] = 0.0f; }
#pragma unroll
    for (int nt = 0; nt < 4; nt++) acco[nt] = vzero8();

    const int NB = NSEQ / 64;
    issue_kv(0, 0);
    for (int kb = 0; kb < NB; ++kb) {
        const int cur = kb & 1;
        if (kb + 1 < NB) {
            issue_kv(kb + 1, cur ^ 1);  // overlap next K/V block behind compute
            if (kb + 2 < NB)
                __builtin_prefetch(K + base + (size_t)(kb + 2) * 64 * DHEAD + tid * 8, 0, 1);
            wait_async_le<8>();         // only block kb's 8 ops must be done
        } else {
            wait_async_le<0>();
        }
        __syncthreads();

        // S = Q * K^T  (16x64 stripe per wave)
        v8f S[4];
#pragma unroll
        for (int ct = 0; ct < 4; ct++) {
            S[ct] = vzero8();
#pragma unroll
            for (int c = 0; c < 2; c++) {
                FragB16 kf;
#pragma unroll
                for (int j = 0; j < 8; j++) {
                    int lk2 = (j & 3) * 2 + (j >> 2) * 16 + half * 8;
                    kf.u[j] = *reinterpret_cast<const unsigned*>(
                        &sK[cur][(ct * 16 + m16) * DHEAD + c * 32 + lk2]);
                }
                S[ct] = __builtin_amdgcn_wmma_f32_16x16x32_bf16(false, qf[c].v, false, kf.v,
                                                                (short)0, S[ct], false, false);
            }
        }

        // online softmax: row j lives in the 16 lanes sharing (lane>>4);
        // reductions stay on the VALU crossbar (DPP), off the DS pipe
#pragma unroll
        for (int j = 0; j < 8; j++) {
            float s = fmaxf(fmaxf(S[0][j], S[1][j]), fmaxf(S[2][j], S[3][j]));
            s = row16_max(s);
            float newm  = fmaxf(mrow[j], s);
            float alpha = __expf(mrow[j] - newm);
            float psum  = 0.0f;
#pragma unroll
            for (int ct = 0; ct < 4; ct++) {
                float p = __expf(S[ct][j] - newm);
                psum += p;
                sP[wave][(j + half * 8) * 64 + ct * 16 + m16] = (bf16_t)p;
            }
            psum    = row16_sum(psum);
            lrow[j] = lrow[j] * alpha + psum;
            mrow[j] = newm;
#pragma unroll
            for (int nt = 0; nt < 4; nt++) acco[nt][j] *= alpha;
        }

        // same-wave LDS RAW: P stored in C-layout, reload in A-layout
        asm volatile("s_wait_dscnt 0x0" ::: "memory");

        FragB16 pf[2];
#pragma unroll
        for (int c = 0; c < 2; c++)
#pragma unroll
            for (int j = 0; j < 8; j++) {
                int kk = c * 32 + (j & 3) * 2 + (j >> 2) * 16 + half * 8;
                pf[c].u[j] = *reinterpret_cast<const unsigned*>(&sP[wave][m16 * 64 + kk]);
            }

        // O += P * V
#pragma unroll
        for (int nt = 0; nt < 4; nt++)
#pragma unroll
            for (int c = 0; c < 2; c++) {
                FragB16 vf;
#pragma unroll
                for (int j = 0; j < 8; j++) {
                    int lk2 = c * 32 + (j & 3) * 2 + (j >> 2) * 16 + half * 8;
                    vf.e[2 * j]     = sV[cur][lk2 * DHEAD + nt * 16 + m16];
                    vf.e[2 * j + 1] = sV[cur][(lk2 + 1) * DHEAD + nt * 16 + m16];
                }
                acco[nt] = __builtin_amdgcn_wmma_f32_16x16x32_bf16(false, pf[c].v, false, vf.v,
                                                                   (short)0, acco[nt], false, false);
            }
        __syncthreads();  // buffer `cur` free for reuse at kb+2
    }

    // normalize and write o as bf16 in [B,N,C] layout
#pragma unroll
    for (int nt = 0; nt < 4; nt++)
#pragma unroll
        for (int j = 0; j < 8; j++) {
            int r     = q0 + j + half * 8;
            float val = acco[nt][j] / lrow[j];
            O[((size_t)(b * NSEQ) + r) * CDIM + h * DHEAD + nt * 16 + m16] = (bf16_t)val;
        }
}

// ---------------------------------------------------------------------------
extern "C" void kernel_launch(void* const* d_in, const int* in_sizes, int n_in,
                              void* d_out, int out_size, void* d_ws, size_t ws_size,
                              hipStream_t stream) {
    (void)in_sizes; (void)n_in; (void)out_size; (void)ws_size;
    const float* x      = (const float*)d_in[0];
    const float* ln1_g  = (const float*)d_in[1];
    const float* ln1_b  = (const float*)d_in[2];
    const float* w_qkv  = (const float*)d_in[3];
    const float* w_proj = (const float*)d_in[4];
    const float* b_proj = (const float*)d_in[5];
    const float* ln2_g  = (const float*)d_in[6];
    const float* ln2_b  = (const float*)d_in[7];
    const float* w_fc1  = (const float*)d_in[8];
    const float* b_fc1  = (const float*)d_in[9];
    const float* w_fc2  = (const float*)d_in[10];
    const float* b_fc2  = (const float*)d_in[11];

    char* base = (char*)d_ws;
    size_t off = 0;
    auto alloc = [&](size_t bytes) -> char* {
        off = (off + 255) & ~(size_t)255;
        char* p = base + off;
        off += bytes;
        return p;
    };
    bf16_t* h1     = (bf16_t*)alloc((size_t)MROWS * CDIM * 2);
    bf16_t* qkv    = (bf16_t*)alloc(3 * QSZ * 2);
    bf16_t* ob     = (bf16_t*)alloc((size_t)MROWS * CDIM * 2);
    float*  x2     = (float*) alloc((size_t)MROWS * CDIM * 4);
    bf16_t* h2     = (bf16_t*)alloc((size_t)MROWS * CDIM * 2);
    bf16_t* mact   = (bf16_t*)alloc((size_t)MROWS * HIDD * 2);
    bf16_t* wqkvb  = (bf16_t*)alloc((size_t)CDIM * 3 * CDIM * 2);
    bf16_t* wprojb = (bf16_t*)alloc((size_t)CDIM * CDIM * 2);
    bf16_t* wfc1b  = (bf16_t*)alloc((size_t)CDIM * HIDD * 2);
    bf16_t* wfc2b  = (bf16_t*)alloc((size_t)HIDD * CDIM * 2);

    int nq = CDIM * 3 * CDIM;
    cvt_kernel<<<(nq + 255) / 256, 256, 0, stream>>>(w_qkv, wqkvb, nq);
    int np = CDIM * CDIM;
    cvt_kernel<<<(np + 255) / 256, 256, 0, stream>>>(w_proj, wprojb, np);
    int n1 = CDIM * HIDD;
    cvt_kernel<<<(n1 + 255) / 256, 256, 0, stream>>>(w_fc1, wfc1b, n1);
    cvt_kernel<<<(n1 + 255) / 256, 256, 0, stream>>>(w_fc2, wfc2b, n1);

    ln_kernel<<<MROWS, 128, 0, stream>>>(x, ln1_g, ln1_b, h1);

    gemm_bf16_kernel<0><<<dim3(3 * CDIM / 128, MROWS / 128), 256, 0, stream>>>(
        h1, wqkvb, 3 * CDIM, CDIM, nullptr, nullptr, nullptr, qkv);

    attn_kernel<<<dim3(B_ * HEADS, NSEQ / 64), 128, 0, stream>>>(
        qkv, qkv + QSZ, qkv + 2 * QSZ, ob);

    gemm_bf16_kernel<1><<<dim3(CDIM / 128, MROWS / 128), 256, 0, stream>>>(
        ob, wprojb, CDIM, CDIM, b_proj, x, x2, nullptr);

    ln_kernel<<<MROWS, 128, 0, stream>>>(x2, ln2_g, ln2_b, h2);

    gemm_bf16_kernel<2><<<dim3(HIDD / 128, MROWS / 128), 256, 0, stream>>>(
        h2, wfc1b, HIDD, CDIM, b_fc1, nullptr, nullptr, mact);

    gemm_bf16_kernel<3><<<dim3(CDIM / 128, MROWS / 128), 256, 0, stream>>>(
        mact, wfc2b, CDIM, HIDD, b_fc2, x2, (float*)d_out, nullptr);
}
